// ConvNorm4ContPit_31052613550086
// MI455X (gfx1250) — matrix operand order repaired and checked by reading, not compiled, not run on hardware
//
#include <hip/hip_runtime.h>

// ---------------- constants ----------------
#define Bsz    32
#define Tsz    1024
#define TP     1028          // T + 2 pad rows each side
#define NUMSEG 54
#define SEGL   64
#define NCAND  (NUMSEG*SEGL) // 3456
#define NGRP   40            // 32 cont groups + 8 pit groups
#define GN_INV (1.0f/16384.0f)
#define EPSG   1e-5f

typedef __bf16 bf16;
typedef __attribute__((ext_vector_type(16))) __bf16 v16bf;
typedef __attribute__((ext_vector_type(8)))  __bf16 v8bf;
typedef __attribute__((ext_vector_type(8)))  float  v8f;
typedef unsigned int v4u __attribute__((ext_vector_type(4)));
typedef int          v8i __attribute__((ext_vector_type(8)));
typedef int          v4i __attribute__((ext_vector_type(4)));

// ---------------- weight repack: [Cout][Cin][5] f32 -> [Cout][5][Ceff] bf16 (ci-padded) ----
__global__ void pack_w_kernel(const float* __restrict__ src, bf16* __restrict__ dst,
                              int Cout, int Cin, int Ceff)
{
    int idx = blockIdx.x * blockDim.x + threadIdx.x;
    int total = Cout * 5 * Ceff;
    if (idx >= total) return;
    int ci = idx % Ceff;
    int k  = (idx / Ceff) % 5;
    int co = idx / (Ceff * 5);
    float v = (ci < Cin) ? src[((size_t)co * Cin + ci) * 5 + k] : 0.0f;
    dst[idx] = (bf16)v;
}

// ---------------- input repack: [B,T,Cin] f32 -> [B,TP,Cs] bf16 at row t+2 ----------------
__global__ void pack_act_kernel(const float* __restrict__ src, bf16* __restrict__ dst,
                                int Cin, int Cs, int total)
{
    int idx = blockIdx.x * blockDim.x + threadIdx.x;
    if (idx >= total) return;
    int c = idx % Cin;
    int t = (idx / Cin) % Tsz;
    int b = idx / (Cin * Tsz);
    dst[((size_t)b * TP + t + 2) * Cs + c] = (bf16)src[idx];
}

// ---------------- TDM helper: 1-row 2D tile of N bf16 elements, global -> LDS ----------------
__device__ __forceinline__ void tdm_load_weights(const bf16* gsrc, void* lds_ptr, unsigned nElems)
{
    unsigned long long ga = (unsigned long long)(uintptr_t)gsrc;
    unsigned lds_addr = (unsigned)(uintptr_t)lds_ptr;   // flat LDS aperture: low 32 bits = LDS offset

    v4u g0;
    g0[0] = 1u;                                          // count=1, is_restore=0, gather off
    g0[1] = lds_addr;                                    // bits 63:32 lds_addr
    g0[2] = (unsigned)ga;                                // global_addr[31:0]
    g0[3] = ((unsigned)(ga >> 32) & 0x01FFFFFFu) | (2u << 30);  // addr[56:32] | type=2

    v8i g1;
    g1[0] = (int)(1u << 16);                             // data_size=1 (2 bytes), mask=0
    g1[1] = (int)((nElems & 0xFFFFu) << 16);             // tensor_dim0[15:0] (atomic addr = 0)
    g1[2] = (int)((nElems >> 16) | (1u << 16));          // tensor_dim0[31:16] | tensor_dim1=1
    g1[3] = (int)((nElems & 0xFFFFu) << 16);             // tensor_dim1[31:16]=0 | tile_dim0
    g1[4] = 1;                                           // tile_dim1=1, tile_dim2=0
    g1[5] = (int)nElems;                                 // tensor_dim0_stride[31:0]
    g1[6] = 0;                                           // stride hi | dim1_stride lo
    g1[7] = 0;                                           // dim1_stride hi

    v4i gz4 = {0, 0, 0, 0};                              // groups 2/3 unused (<=2D tensor)
    v8i gz8 = {0, 0, 0, 0, 0, 0, 0, 0};                  // extra descriptor group (unused)

    __builtin_amdgcn_tensor_load_to_lds(g0, g1, gz4, gz4, gz8, 0);
    __builtin_amdgcn_s_wait_tensorcnt(0);
}

// ---------------- conv1d as implicit GEMM via WMMA bf16 ----------------
// act   : bf16 [B, TP, Cs]   (time-major, 2 zero pad rows each side)
// wpack : bf16 [Cout][5][CEFF]
// out   : f32  [B, 1024, 640] at column base `cobase`
// stats : f32  [B, NGRP, 2]  (sum, sumsq), atomic accumulate
// block = 256 threads (8 waves); tile = 16 Cout (one GN group) x 128 t
// weight tile for the block (16 x 5 x CEFF bf16) staged into LDS via TDM
template<int CEFF>
__global__ __launch_bounds__(256) void conv_wmma_kernel(
    const bf16* __restrict__ act, int Cs,
    const bf16* __restrict__ wpack,
    const float* __restrict__ bias,
    float* __restrict__ mfout, int cobase,
    float* __restrict__ stats, int gOffset)
{
    extern __shared__ char dynsm[];
    bf16* smemW = (bf16*)dynsm;                      // [16][5][CEFF]

    const int b    = blockIdx.z;
    const int coT  = blockIdx.y;
    const int t0   = blockIdx.x * 128;
    const int co0  = coT * 16;
    const int wave = threadIdx.x >> 5;
    const int lane = threadIdx.x & 31;
    const int t0w  = t0 + wave * 16;
    const int nL   = lane & 15;
    const int hi   = lane >> 4;

    // ---- stage the block's weight tile into LDS with the Tensor Data Mover ----
    if (threadIdx.x < 32) {
        tdm_load_weights(wpack + (size_t)co0 * 5 * CEFF, smemW, 16u * 5u * CEFF);
    }
    __syncthreads();

    v8f acc = {0.f, 0.f, 0.f, 0.f, 0.f, 0.f, 0.f, 0.f};

    const bf16* actB = act + (size_t)b * TP * Cs;
    const bf16* wRow = smemW + (size_t)nL * 5 * CEFF;    // A row for M = nL (in LDS)

#pragma unroll
    for (int k = 0; k < 5; ++k) {
        const bf16* arow = wRow + (size_t)k * CEFF;
        const bf16* brow = actB + (size_t)(t0w + nL + k) * Cs + hi * 16;
        if (k < 4) __builtin_prefetch(brow + Cs, 0, 3);   // next shift's row
#pragma unroll 4
        for (int ci0 = 0; ci0 < CEFF; ci0 += 32) {
            v16bf afrag = *(const v16bf*)(arow + ci0 + hi * 16);
            v16bf bfrag = *(const v16bf*)(brow + ci0);
            acc = __builtin_amdgcn_wmma_f32_16x16x32_bf16(
                false, afrag, false, bfrag, (short)0, acc, false, false);
        }
    }

    // bias (acc[r] -> co = co0 + hi*8 + r)
#pragma unroll
    for (int r = 0; r < 8; ++r) acc[r] += bias[co0 + hi * 8 + r];

    // store: lane holds t = t0w + nL, cols co0 + hi*8 .. +7 (contiguous)
    {
        const int trow = t0w + nL;
        float* orow = mfout + ((size_t)b * Tsz + trow) * 640 + cobase + co0 + hi * 8;
        *(float4*)(orow)     = make_float4(acc[0], acc[1], acc[2], acc[3]);
        *(float4*)(orow + 4) = make_float4(acc[4], acc[5], acc[6], acc[7]);
    }

    // block reduction of sum / sumsq for GroupNorm stats
    __shared__ float rsum[256];
    __shared__ float rsq[256];
    float s = 0.f, s2 = 0.f;
#pragma unroll
    for (int i = 0; i < 8; ++i) { float v = acc[i]; s += v; s2 += v * v; }
    rsum[threadIdx.x] = s;
    rsq[threadIdx.x]  = s2;
    __syncthreads();
    for (int st = 128; st > 0; st >>= 1) {
        if ((int)threadIdx.x < st) {
            rsum[threadIdx.x] += rsum[threadIdx.x + st];
            rsq[threadIdx.x]  += rsq[threadIdx.x + st];
        }
        __syncthreads();
    }
    if (threadIdx.x == 0) {
        float* sp = stats + ((size_t)b * NGRP + gOffset + coT) * 2;
        atomicAdd(sp,     rsum[0]);
        atomicAdd(sp + 1, rsq[0]);
    }
}

// ---------------- GroupNorm + ReLU (in place on mfraw [B,1024,640]) ----------------
__global__ __launch_bounds__(160) void gn_relu_kernel(
    float* __restrict__ mf, const float* __restrict__ stats,
    const float* __restrict__ cg, const float* __restrict__ cbeta,
    const float* __restrict__ pg, const float* __restrict__ pbeta)
{
    int row = blockIdx.x;
    int b = row >> 10;
    int c = threadIdx.x * 4;
    float* p = mf + (size_t)row * 640 + c;
    float4 x = *(float4*)p;
    int g = (c < 512) ? (c >> 4) : 32 + ((c - 512) >> 4);
    const float* sp = stats + ((size_t)b * NGRP + g) * 2;
    float mean = sp[0] * GN_INV;
    float var  = sp[1] * GN_INV - mean * mean;
    float inv  = rsqrtf(var + EPSG);
    float y[4] = {x.x, x.y, x.z, x.w};
#pragma unroll
    for (int e = 0; e < 4; ++e) {
        int cc = c + e;
        float gm, bt;
        if (cc < 512) { gm = cg[cc]; bt = cbeta[cc]; }
        else          { gm = pg[cc - 512]; bt = pbeta[cc - 512]; }
        y[e] = fmaxf((y[e] - mean) * inv * gm + bt, 0.0f);
    }
    *(float4*)p = make_float4(y[0], y[1], y[2], y[3]);
}

// ---------------- random resample: scan + compacting lerp scatter ----------------
__global__ __launch_bounds__(256) void resample_kernel(
    const float* __restrict__ mf,      // [B,1024,640] normalized
    const float* __restrict__ scales,  // layer slice [B*NUMSEG]
    const int*   __restrict__ lens,    // layer slice [B*NUMSEG]
    bf16* __restrict__ contA,          // [B,TP,512] (pre-zeroed) -- when !finalFlag
    bf16* __restrict__ pitA,           // [B,TP,128]
    float* __restrict__ outMel,        // [B,1024,512] -- when finalFlag
    float* __restrict__ outF0,         // [B,1024,128]
    int finalFlag)
{
    const int b = blockIdx.x;
    __shared__ int   s_offs[NUMSEG];
    __shared__ int   s_dst[NCAND];
    __shared__ int   s_gi[NCAND];
    __shared__ float s_lam[NCAND];
    __shared__ int   s_tot[256];

    if (threadIdx.x == 0) {
        int a = 0;
        for (int s = 0; s < NUMSEG; ++s) { s_offs[s] = a; a += lens[b * NUMSEG + s]; }
    }
    __syncthreads();

    const int CPT = 14;                 // 256*14 >= 3456
    int j0 = threadIdx.x * CPT;
    unsigned mbits = 0;
    int cnt = 0;
    for (int jj = 0; jj < CPT; ++jj) {
        int j = j0 + jj;
        if (j >= NCAND) break;
        int s = j >> 6;
        float fi = (float)(j & 63);
        float scale = scales[b * NUMSEG + s];
        float is = fi / scale;
        float fl = floorf(is);
        int len = lens[b * NUMSEG + s];
        int iorg = (int)fl + s_offs[s];
        bool m = (fl < (float)(len - 1)) && (iorg < Tsz - 1);
        if (m) { mbits |= (1u << jj); cnt++; }
        s_gi[j]  = min(max(iorg, 0), Tsz - 2);
        s_lam[j] = is - fl;
    }
    s_tot[threadIdx.x] = cnt;
    __syncthreads();
    if (threadIdx.x == 0) {
        int a = 0;
        for (int t = 0; t < 256; ++t) { int c = s_tot[t]; s_tot[t] = a; a += c; }
    }
    __syncthreads();
    int base = s_tot[threadIdx.x];
    for (int jj = 0; jj < CPT; ++jj) {
        int j = j0 + jj;
        if (j >= NCAND) break;
        if (mbits & (1u << jj)) { s_dst[j] = (base < Tsz) ? base : -1; base++; }
        else                    { s_dst[j] = -1; }
    }
    __syncthreads();

    // copy phase: wave per candidate row, lane covers 20 channels (5 x float4)
    const int wave = threadIdx.x >> 5;
    const int lane = threadIdx.x & 31;
    for (int j = wave; j < NCAND; j += 8) {
        int d = s_dst[j];
        if (d < 0) continue;
        float lam = s_lam[j];
        int gi = s_gi[j];
        const float* r0 = mf + ((size_t)b * Tsz + gi) * 640;
        const float* r1 = r0 + 640;
        int cbase = lane * 20;
#pragma unroll
        for (int u = 0; u < 5; ++u) {
            int c0 = cbase + 4 * u;
            float4 x0 = *(const float4*)(r0 + c0);
            float4 x1 = *(const float4*)(r1 + c0);
            float y[4];
            y[0] = x0.x + lam * (x1.x - x0.x);
            y[1] = x0.y + lam * (x1.y - x0.y);
            y[2] = x0.z + lam * (x1.z - x0.z);
            y[3] = x0.w + lam * (x1.w - x0.w);
            if (finalFlag) {
#pragma unroll
                for (int e = 0; e < 4; ++e) {
                    int cc = c0 + e;
                    if (cc < 512) outMel[((size_t)b * Tsz + d) * 512 + cc] = y[e];
                    else          outF0 [((size_t)b * Tsz + d) * 128 + (cc - 512)] = y[e];
                }
            } else {
#pragma unroll
                for (int e = 0; e < 4; ++e) {
                    int cc = c0 + e;
                    if (cc < 512) contA[((size_t)b * TP + d + 2) * 512 + cc] = (bf16)y[e];
                    else          pitA [((size_t)b * TP + d + 2) * 128 + (cc - 512)] = (bf16)y[e];
                }
            }
        }
    }
}

// ---------------- host orchestration ----------------
extern "C" void kernel_launch(void* const* d_in, const int* in_sizes, int n_in,
                              void* d_out, int out_size, void* d_ws, size_t ws_size,
                              hipStream_t stream)
{
    (void)in_sizes; (void)n_in; (void)ws_size;

    const float* mel   = (const float*)d_in[0];
    const float* f0    = (const float*)d_in[1];
    const float* cw0   = (const float*)d_in[2];
    const float* cw    = (const float*)d_in[3];
    const float* cb    = (const float*)d_in[4];
    const float* cg    = (const float*)d_in[5];
    const float* cbeta = (const float*)d_in[6];
    const float* pw0   = (const float*)d_in[7];
    const float* pw    = (const float*)d_in[8];
    const float* pb    = (const float*)d_in[9];
    const float* pg    = (const float*)d_in[10];
    const float* pbeta = (const float*)d_in[11];
    const float* scales= (const float*)d_in[12];
    const int*   lens  = (const int*)d_in[13];

    char* ws = (char*)d_ws;
    size_t off = 0;
    auto take = [&](size_t bytes) -> void* {
        void* p = ws + off;
        off = (off + bytes + 255) & ~(size_t)255;
        return p;
    };
    const size_t contActBytes = (size_t)Bsz * TP * 512 * sizeof(bf16);
    const size_t pitActBytes  = (size_t)Bsz * TP * 128 * sizeof(bf16);
    const size_t mfrawBytes   = (size_t)Bsz * Tsz * 640 * sizeof(float);
    const size_t statsBytes   = (size_t)Bsz * NGRP * 2 * sizeof(float);
    const size_t wcElems      = (size_t)512*5*128 + 2*(size_t)512*5*512;
    const size_t wpElems      = (size_t)128*5*32  + 2*(size_t)128*5*128;

    bf16*  contAct = (bf16*) take(contActBytes);
    bf16*  pitAct  = (bf16*) take(pitActBytes);
    float* mfraw   = (float*)take(mfrawBytes);
    float* stats   = (float*)take(statsBytes);
    bf16*  wc      = (bf16*) take(wcElems * sizeof(bf16));
    bf16*  wp      = (bf16*) take(wpElems * sizeof(bf16));

    const size_t WC1 = (size_t)512*5*128, WC2 = WC1 + (size_t)512*5*512;
    const size_t WP1 = (size_t)128*5*32,  WP2 = WP1 + (size_t)128*5*128;

    // ---- weight repack (bf16) ----
    {
        int tw;
        tw = 512*5*128; pack_w_kernel<<<(tw+255)/256,256,0,stream>>>(cw0, wc,       512,  80, 128);
        tw = 512*5*512; pack_w_kernel<<<(tw+255)/256,256,0,stream>>>(cw,  wc + WC1, 512, 512, 512);
                        pack_w_kernel<<<(tw+255)/256,256,0,stream>>>(cw + (size_t)512*512*5, wc + WC2, 512, 512, 512);
        tw = 128*5*32;  pack_w_kernel<<<(tw+255)/256,256,0,stream>>>(pw0, wp,       128,   4,  32);
        tw = 128*5*128; pack_w_kernel<<<(tw+255)/256,256,0,stream>>>(pw,  wp + WP1, 128, 128, 128);
                        pack_w_kernel<<<(tw+255)/256,256,0,stream>>>(pw + (size_t)128*128*5, wp + WP2, 128, 128, 128);
    }

    // ---- input repack (zero pads + unused channels) ----
    (void)hipMemsetAsync(contAct, 0, contActBytes, stream);
    (void)hipMemsetAsync(pitAct,  0, pitActBytes,  stream);
    {
        int tm = Bsz*Tsz*80; pack_act_kernel<<<(tm+255)/256,256,0,stream>>>(mel, contAct, 80, 512, tm);
        int tf = Bsz*Tsz*4;  pack_act_kernel<<<(tf+255)/256,256,0,stream>>>(f0,  pitAct,   4, 128, tf);
    }

    float* outMel = (float*)d_out;
    float* outF0  = outMel + (size_t)Bsz * Tsz * 512;

    for (int l = 0; l < 3; ++l) {
        (void)hipMemsetAsync(stats, 0, statsBytes, stream);

        const bf16* wcL = wc + (l == 0 ? 0 : (l == 1 ? WC1 : WC2));
        const bf16* wpL = wp + (l == 0 ? 0 : (l == 1 ? WP1 : WP2));

        // cont conv: 512 out channels
        if (l == 0) {
            conv_wmma_kernel<128><<<dim3(8, 32, Bsz), 256, 16*5*128*sizeof(bf16), stream>>>(
                contAct, 512, wcL, cb + l*512, mfraw, 0, stats, 0);
        } else {
            conv_wmma_kernel<512><<<dim3(8, 32, Bsz), 256, 16*5*512*sizeof(bf16), stream>>>(
                contAct, 512, wcL, cb + l*512, mfraw, 0, stats, 0);
        }
        // pit conv: 128 out channels
        if (l == 0) {
            conv_wmma_kernel<32><<<dim3(8, 8, Bsz), 256, 16*5*32*sizeof(bf16), stream>>>(
                pitAct, 128, wpL, pb + l*128, mfraw, 512, stats, 32);
        } else {
            conv_wmma_kernel<128><<<dim3(8, 8, Bsz), 256, 16*5*128*sizeof(bf16), stream>>>(
                pitAct, 128, wpL, pb + l*128, mfraw, 512, stats, 32);
        }

        gn_relu_kernel<<<Bsz*Tsz, 160, 0, stream>>>(
            mfraw, stats, cg + l*512, cbeta + l*512, pg + l*128, pbeta + l*128);

        if (l < 2) {
            (void)hipMemsetAsync(contAct, 0, contActBytes, stream);
            (void)hipMemsetAsync(pitAct,  0, pitActBytes,  stream);
        } else {
            (void)hipMemsetAsync(d_out, 0, (size_t)out_size * sizeof(float), stream);
        }

        resample_kernel<<<Bsz, 256, 0, stream>>>(
            mfraw, scales + (size_t)l * Bsz * NUMSEG, lens + (size_t)l * Bsz * NUMSEG,
            contAct, pitAct, outMel, outF0, (l == 2) ? 1 : 0);
    }
}
